// MolInteraction_87978110091590
// MI455X (gfx1250) — matrix operand order repaired and checked by reading, not compiled, not run on hardware
//
#include <hip/hip_runtime.h>
#include <hip/hip_bf16.h>

typedef __attribute__((ext_vector_type(16))) _Float16 v16h;
typedef __attribute__((ext_vector_type(8)))  _Float16 v8h;
typedef __attribute__((ext_vector_type(4)))  _Float16 v4h;
typedef __attribute__((ext_vector_type(8)))  float    v8f;

#define D_ 128
#define H_ 256

// ---------------- fragment loaders (CDNA5 v_wmma_f32_16x16x32_f16 layouts) -------------
// A (16x32 f16): lanes 0-15 -> row M=lane, halves[0..7]=K0..7, halves[8..15]=K16..23
//                lanes16-31 -> row M=lane-16, halves = K8..15, K24..31
__device__ __forceinline__ v16h frag_a(const _Float16* base, int stride, int lane) {
    const _Float16* p = base + (lane & 15) * stride + ((lane >> 4) * 8);
    v8h lo = *(const v8h*)(p);
    v8h hi = *(const v8h*)(p + 16);
    v16h f;
#pragma unroll
    for (int i = 0; i < 8; ++i) { f[i] = lo[i]; f[8 + i] = hi[i]; }
    return f;
}
// B (32x16 f16, W stored transposed so "row" = output column):
// lanes 0-15 -> col N=lane, halves = K0..7, K16..23; lanes16-31 -> K8..15, K24..31
__device__ __forceinline__ v16h frag_b(const _Float16* base, int stride, int lane) {
    const _Float16* p = base + (lane & 15) * stride + ((lane >> 4) * 8);
    v8h lo = *(const v8h*)(p);
    v8h hi = *(const v8h*)(p + 16);
    v16h f;
#pragma unroll
    for (int i = 0; i < 8; ++i) { f[i] = lo[i]; f[8 + i] = hi[i]; }
    return f;
}

__device__ __forceinline__ float sigm(float x) { return 1.0f / (1.0f + __expf(-x)); }

__device__ __forceinline__ v8f wmma_f16(v16h a, v16h b, v8f c) {
    return __builtin_amdgcn_wmma_f32_16x16x32_f16(false, a, false, b, (short)0, c, false, false);
}

// ------ gated-MLP core, 32-row M tile: X(32xIN f16 LDS) -> four 16x16 f32 tiles/wave ---
// GEMM1: 2 branches x 16 H-tiles = 32 jobs over 8 waves (4 each); each B fragment feeds
//        2 WMMAs (M-tiles 0/1). relu -> LDS hidden f16 [branch][32][H].
// GEMM2: wave w computes output cols w*16..w*16+15 for BOTH branches, both M-tiles.
template<int IN>
__device__ __forceinline__ void gated_core32(
    const _Float16* sX, _Float16* sH,
    const _Float16* __restrict__ W1t, const float* __restrict__ b1,
    const _Float16* __restrict__ W2t, const float* __restrict__ b2,
    v8f o0[2], v8f o1[2])
{
    const int t = threadIdx.x;
    const int lane = t & 31;
    const int wave = t >> 5;
    const v8f vz = {0.f,0.f,0.f,0.f,0.f,0.f,0.f,0.f};
    v8f acc[4][2];
#pragma unroll
    for (int j = 0; j < 4; ++j) { acc[j][0] = vz; acc[j][1] = vz; }

    for (int ks = 0; ks < IN / 32; ++ks) {
        v16h a0 = frag_a(sX + ks * 32, IN, lane);
        v16h a1 = frag_a(sX + 16 * IN + ks * 32, IN, lane);
#pragma unroll
        for (int j = 0; j < 4; ++j) {
            const int job = wave + 8 * j;
            const int br = job >> 4, nt = job & 15;
            v16h bf = frag_b(W1t + (br * H_ + nt * 16) * IN + ks * 32, IN, lane);
            acc[j][0] = wmma_f16(a0, bf, acc[j][0]);
            acc[j][1] = wmma_f16(a1, bf, acc[j][1]);
        }
    }
    const int rb = (lane >> 4) * 8;
#pragma unroll
    for (int j = 0; j < 4; ++j) {
        const int job = wave + 8 * j;
        const int br = job >> 4, nt = job & 15;
        const int col = nt * 16 + (lane & 15);
        const float bias = b1[br * H_ + col];
#pragma unroll
        for (int mt = 0; mt < 2; ++mt) {
#pragma unroll
            for (int r = 0; r < 8; ++r) {
                float v = acc[j][mt][r] + bias;
                sH[br * 32 * H_ + (mt * 16 + rb + r) * H_ + col] = (_Float16)fmaxf(v, 0.f);
            }
        }
    }
    __syncthreads();
    v8f c0[2] = {vz, vz}, c1[2] = {vz, vz};
    for (int ks = 0; ks < H_ / 32; ++ks) {
        v16h x00 = frag_a(sH + ks * 32, H_, lane);            // br0, mtile0
        v16h x01 = frag_a(sH + 16 * H_ + ks * 32, H_, lane);  // br0, mtile1
        v16h x10 = frag_a(sH + 32 * H_ + ks * 32, H_, lane);  // br1, mtile0
        v16h x11 = frag_a(sH + 48 * H_ + ks * 32, H_, lane);  // br1, mtile1
        v16h w0 = frag_b(W2t + (0 * D_ + wave * 16) * H_ + ks * 32, H_, lane);
        v16h w1 = frag_b(W2t + (1 * D_ + wave * 16) * H_ + ks * 32, H_, lane);
        c0[0] = wmma_f16(x00, w0, c0[0]);
        c0[1] = wmma_f16(x01, w0, c0[1]);
        c1[0] = wmma_f16(x10, w1, c1[0]);
        c1[1] = wmma_f16(x11, w1, c1[1]);
    }
    o0[0] = c0[0]; o0[1] = c0[1]; o1[0] = c1[0]; o1[1] = c1[1];
}

// ------------------------------- AtomConv MLP (32 rows/block) --------------------------
__global__ __launch_bounds__(256) void atomconv_kernel(
    const float* __restrict__ atom_feat, const float* __restrict__ bond_feat,
    const float* __restrict__ aew,
    const int* __restrict__ src_a, const int* __restrict__ dst_a,
    const _Float16* __restrict__ W1t, const float* __restrict__ b1,
    const _Float16* __restrict__ W2t, const float* __restrict__ b2,
    float* __restrict__ h_a)
{
    constexpr int IN = 3 * D_;
    __shared__ _Float16 sX[32 * IN];
    __shared__ _Float16 sH[2 * 32 * H_];
    __shared__ int sI[2][32];
    const int e0 = blockIdx.x * 32;
    const int t = threadIdx.x;
    if (t < 64) sI[t >> 5][t & 31] = (t < 32) ? src_a[e0 + t] : dst_a[e0 + (t & 31)];
    __syncthreads();
    for (int i = t * 4; i < 32 * IN; i += 1024) {
        const int m = i / IN, k = i - m * IN;
        float4 v;
        if (k < D_)            v = *(const float4*)&atom_feat[(size_t)sI[0][m] * D_ + k];
        else if (k < 2 * D_)   v = *(const float4*)&atom_feat[(size_t)sI[1][m] * D_ + (k - D_)];
        else                   v = *(const float4*)&bond_feat[(size_t)(e0 + m) * D_ + (k - 2 * D_)];
        v4h h4 = {(_Float16)v.x, (_Float16)v.y, (_Float16)v.z, (_Float16)v.w};
        *(v4h*)&sX[i] = h4;
    }
    __syncthreads();
    v8f o0[2], o1[2];
    gated_core32<IN>(sX, sH, W1t, b1, W2t, b2, o0, o1);
    const int lane = t & 31, wave = t >> 5;
    const int col = wave * 16 + (lane & 15);
    const int rb = (lane >> 4) * 8;
    const float c0 = b2[col], c1 = b2[D_ + col];
#pragma unroll
    for (int mt = 0; mt < 2; ++mt) {
#pragma unroll
        for (int r = 0; r < 8; ++r) {
            const int m = mt * 16 + rb + r;
            const float y0 = o0[mt][r] + c0, y1 = o1[mt][r] + c1;
            const float v = y0 * sigm(y0) * sigm(y1) * aew[(size_t)(e0 + m) * D_ + col];
            atomicAdd(&h_a[(size_t)sI[1][m] * D_ + col], v);
        }
    }
}

// --------------- BondConv (EPI=0) / AngleUpdate (EPI=1) MLP, 32 rows/block -------------
template<int EPI>
__global__ __launch_bounds__(256) void edge_mlp_kernel(
    const float* __restrict__ f2,         // bond_feat (EPI=0) or bond_out (EPI=1)
    const float* __restrict__ angle_feat,
    const float* __restrict__ atom_out,
    const float* __restrict__ bw,         // only EPI=0
    const int* __restrict__ src_b, const int* __restrict__ dst_b,
    const int* __restrict__ angle_index,
    const _Float16* __restrict__ W1t, const float* __restrict__ b1,
    const _Float16* __restrict__ W2t, const float* __restrict__ b2,
    float* __restrict__ out)              // h_b (EPI=0) or angle_out (EPI=1)
{
    constexpr int IN = 4 * D_;
    __shared__ _Float16 sX[32 * IN];
    __shared__ _Float16 sH[2 * 32 * H_];
    __shared__ int sI[3][32];
    const int e0 = blockIdx.x * 32;
    const int t = threadIdx.x;
    if (t < 96) {
        const int g = t >> 5, m = t & 31;
        sI[g][m] = (g == 0) ? src_b[e0 + m]
                 : (g == 1) ? dst_b[e0 + m]
                            : angle_index[(e0 + m) * 3 + 1];
    }
    __syncthreads();
    for (int i = t * 4; i < 32 * IN; i += 1024) {
        const int m = i / IN, k = i - m * IN;
        float4 v;
        if (k < D_)            v = *(const float4*)&f2[(size_t)sI[0][m] * D_ + k];
        else if (k < 2 * D_)   v = *(const float4*)&f2[(size_t)sI[1][m] * D_ + (k - D_)];
        else if (k < 3 * D_)   v = *(const float4*)&angle_feat[(size_t)(e0 + m) * D_ + (k - 2 * D_)];
        else                   v = *(const float4*)&atom_out[(size_t)sI[2][m] * D_ + (k - 3 * D_)];
        v4h h4 = {(_Float16)v.x, (_Float16)v.y, (_Float16)v.z, (_Float16)v.w};
        *(v4h*)&sX[i] = h4;
    }
    __syncthreads();
    v8f o0[2], o1[2];
    gated_core32<IN>(sX, sH, W1t, b1, W2t, b2, o0, o1);
    const int lane = t & 31, wave = t >> 5;
    const int col = wave * 16 + (lane & 15);
    const int rb = (lane >> 4) * 8;
    const float c0 = b2[col], c1 = b2[D_ + col];
#pragma unroll
    for (int mt = 0; mt < 2; ++mt) {
#pragma unroll
        for (int r = 0; r < 8; ++r) {
            const int m = mt * 16 + rb + r;
            const float y0 = o0[mt][r] + c0, y1 = o1[mt][r] + c1;
            float v = y0 * sigm(y0) * sigm(y1);
            if (EPI == 0) {
                v *= bw[(size_t)sI[0][m] * D_ + col] * bw[(size_t)sI[1][m] * D_ + col];
                atomicAdd(&out[(size_t)sI[1][m] * D_ + col], v);
            } else {
                const size_t e = (size_t)(e0 + m);
                out[e * D_ + col] = angle_feat[e * D_ + col] + v;
            }
        }
    }
}

// ------------------- residual linear: out = feat + h @ W + b (WMMA) --------------------
__global__ __launch_bounds__(256) void linres_kernel(
    const float* __restrict__ h, const float* __restrict__ feat,
    const _Float16* __restrict__ Wt, const float* __restrict__ b,
    float* __restrict__ out)
{
    __shared__ _Float16 sX[16 * D_];
    const int r0 = blockIdx.x * 16;
    const int t = threadIdx.x;
    for (int i = t * 4; i < 16 * D_; i += 1024) {
        float4 v = *(const float4*)&h[(size_t)r0 * D_ + i];
        v4h h4 = {(_Float16)v.x, (_Float16)v.y, (_Float16)v.z, (_Float16)v.w};
        *(v4h*)&sX[i] = h4;
    }
    __syncthreads();
    const int lane = t & 31, wave = t >> 5;
    v8f acc = {0.f,0.f,0.f,0.f,0.f,0.f,0.f,0.f};
#pragma unroll
    for (int ks = 0; ks < D_ / 32; ++ks) {
        v16h a = frag_a(sX + ks * 32, D_, lane);
        v16h w = frag_b(Wt + (wave * 16) * D_ + ks * 32, D_, lane);
        acc = wmma_f16(a, w, acc);
    }
    const int col = wave * 16 + (lane & 15);
    const int rb = (lane >> 4) * 8;
    const float bb = b[col];
#pragma unroll
    for (int r = 0; r < 8; ++r) {
        const size_t m = (size_t)(r0 + rb + r);
        out[m * D_ + col] = feat[m * D_ + col] + acc[r] + bb;
    }
}

// -------------------------------- utility kernels --------------------------------------
// dst[b][c][r] = (f16) src[b][r][c]  (weights stored transposed, f16)
__global__ void transpose_f16_kernel(const float* __restrict__ src, _Float16* __restrict__ dst,
                                     int R, int C, int B)
{
    const int i = blockIdx.x * 256 + threadIdx.x;
    const int tot = R * C * B;
    if (i >= tot) return;
    const int bb = i / (R * C);
    const int rem = i - bb * R * C;
    const int r = rem / C, c = rem - r * C;
    dst[bb * R * C + c * R + r] = (_Float16)src[i];
}

__global__ void zero_kernel(float* __restrict__ p, int n)
{
    for (int i = blockIdx.x * 256 + threadIdx.x; i < n; i += gridDim.x * 256) p[i] = 0.f;
}

// ----------------------------------- launcher ------------------------------------------
extern "C" void kernel_launch(void* const* d_in, const int* in_sizes, int n_in,
                              void* d_out, int out_size, void* d_ws, size_t ws_size,
                              hipStream_t stream) {
    constexpr int N_A = 10000, E_A = 160000, E_B = 320000;

    const float* atom_feat  = (const float*)d_in[0];
    const float* bond_feat  = (const float*)d_in[1];
    const float* angle_feat = (const float*)d_in[2];
    const float* aew        = (const float*)d_in[3];
    const float* bw         = (const float*)d_in[4];
    const float* agW1 = (const float*)d_in[5];
    const float* agb1 = (const float*)d_in[6];
    const float* agW2 = (const float*)d_in[7];
    const float* agb2 = (const float*)d_in[8];
    const float* bgW1 = (const float*)d_in[9];
    const float* bgb1 = (const float*)d_in[10];
    const float* bgW2 = (const float*)d_in[11];
    const float* bgb2 = (const float*)d_in[12];
    const float* ngW1 = (const float*)d_in[13];
    const float* ngb1 = (const float*)d_in[14];
    const float* ngW2 = (const float*)d_in[15];
    const float* ngb2 = (const float*)d_in[16];
    const float* linA_W = (const float*)d_in[17];
    const float* linA_b = (const float*)d_in[18];
    const float* linB_W = (const float*)d_in[19];
    const float* linB_b = (const float*)d_in[20];
    const int* src_a = (const int*)d_in[21];
    const int* dst_a = (const int*)d_in[22];
    const int* src_b = (const int*)d_in[23];
    const int* dst_b = (const int*)d_in[24];
    const int* angle_index = (const int*)d_in[25];

    float* atom_out  = (float*)d_out;
    float* bond_out  = atom_out + (size_t)N_A * D_;
    float* angle_out = bond_out + (size_t)E_A * D_;
    // scratch accumulators live in yet-unwritten d_out regions:
    float* h_a = bond_out;   // consumed by linA before linB overwrites region
    float* h_b = angle_out;  // consumed by linB before angle kernel overwrites region

    // f16 transposed weights in d_ws (~1.9 MB)
    _Float16* agW1t = (_Float16*)d_ws;
    _Float16* agW2t = agW1t + 2 * 256 * 384;
    _Float16* bgW1t = agW2t + 2 * 128 * 256;
    _Float16* bgW2t = bgW1t + 2 * 256 * 512;
    _Float16* ngW1t = bgW2t + 2 * 128 * 256;
    _Float16* ngW2t = ngW1t + 2 * 256 * 512;
    _Float16* linAt = ngW2t + 2 * 128 * 256;
    _Float16* linBt = linAt + 128 * 128;

    auto T = [&](const float* s, _Float16* d, int R, int C, int B) {
        const int tot = R * C * B;
        transpose_f16_kernel<<<(tot + 255) / 256, 256, 0, stream>>>(s, d, R, C, B);
    };
    T(agW1, agW1t, 384, 256, 2); T(agW2, agW2t, 256, 128, 2);
    T(bgW1, bgW1t, 512, 256, 2); T(bgW2, bgW2t, 256, 128, 2);
    T(ngW1, ngW1t, 512, 256, 2); T(ngW2, ngW2t, 256, 128, 2);
    T(linA_W, linAt, 128, 128, 1); T(linB_W, linBt, 128, 128, 1);

    zero_kernel<<<2048, 256, 0, stream>>>(h_a, N_A * D_);
    zero_kernel<<<8192, 256, 0, stream>>>(h_b, E_A * D_);

    atomconv_kernel<<<E_A / 32, 256, 0, stream>>>(
        atom_feat, bond_feat, aew, src_a, dst_a, agW1t, agb1, agW2t, agb2, h_a);

    linres_kernel<<<N_A / 16, 256, 0, stream>>>(h_a, atom_feat, linAt, linA_b, atom_out);

    edge_mlp_kernel<0><<<E_B / 32, 256, 0, stream>>>(
        bond_feat, angle_feat, atom_out, bw, src_b, dst_b, angle_index,
        bgW1t, bgb1, bgW2t, bgb2, h_b);

    linres_kernel<<<E_A / 16, 256, 0, stream>>>(h_b, bond_feat, linBt, linB_b, bond_out);

    edge_mlp_kernel<1><<<E_B / 32, 256, 0, stream>>>(
        bond_out, angle_feat, atom_out, nullptr, src_b, dst_b, angle_index,
        ngW1t, ngb1, ngW2t, ngb2, angle_out);
}
